// OlmoLMHeadCE_13451837571432
// MI455X (gfx1250) — matrix operand order repaired and checked by reading, not compiled, not run on hardware
//
#include <hip/hip_runtime.h>
#include <hip/hip_bf16.h>

// Problem constants (match reference)
#define BT_     2048
#define D_      4096
#define V_      100352
#define BM      64              // tokens per block (2 M-strips per wave)
#define BN      256             // vocab columns per block
#define BK      32              // K step (bf16 WMMA K)
#define NCHUNK  (V_ / BN)       // 392
#define MTILES  (BT_ / BM)      // 32
#define KSTEPS  (D_ / BK)       // 128
#define KP      40              // padded LDS K stride (ushorts)
#define IGNORE_INDEX (-100)
#define Z_MULT  1e-4f

typedef __attribute__((ext_vector_type(16))) __bf16          v16bf;
typedef __attribute__((ext_vector_type(8)))  float           v8f;
typedef __attribute__((ext_vector_type(8)))  unsigned short  ushort8;

union BfPack { v16bf bf; ushort8 h[2]; };

// Truncating hi/lo split: r = f - trunc_bf16(f) is EXACT (low mantissa bits),
// so hi+lo carries ~16 mantissa bits at ~4 VALU ops per value.
__device__ __forceinline__ void split_bf16(float f, unsigned& hi, unsigned& lo) {
  const unsigned u = __float_as_uint(f);
  hi = u >> 16;
  const float r = f - __uint_as_float(u & 0xFFFF0000u);
  lo = __float_as_uint(r) >> 16;
}

// Online-softmax partials for one 16x64 strip of logits held in 4 WMMA accs.
__device__ __forceinline__ void strip_epilogue(v8f c0, v8f c1, v8f c2, v8f c3,
    int msub, int nq, int lhalf, int l16, int m0, int n0,
    const int* __restrict__ labels,
    float red_m[][4], float red_s[][4], float* __restrict__ tgt)
{
#pragma unroll
  for (int g = 0; g < 8; ++g) {
    const float a0 = c0[g], a1 = c1[g], a2 = c2[g], a3 = c3[g];
    float mx = fmaxf(fmaxf(a0, a1), fmaxf(a2, a3));
#pragma unroll
    for (int off = 1; off < 16; off <<= 1) mx = fmaxf(mx, __shfl_xor(mx, off, 32));
    float s = __expf(a0 - mx) + __expf(a1 - mx) + __expf(a2 - mx) + __expf(a3 - mx);
#pragma unroll
    for (int off = 1; off < 16; off <<= 1) s += __shfl_xor(s, off, 32);
    if (l16 == g) {
      const int rl = msub * 16 + g + 8 * lhalf;
      red_m[rl][nq] = mx;
      red_s[rl][nq] = s;
    }
    const int row_tok = m0 + msub * 16 + g + 8 * lhalf;
    const int lbl = labels[row_tok];
#pragma unroll
    for (int j = 0; j < 4; ++j) {
      const int col = n0 + nq * 64 + j * 16 + l16;
      if (lbl == col)
        tgt[row_tok] = ((j == 0) ? c0 : (j == 1) ? c1 : (j == 2) ? c2 : c3)[g];
    }
  }
}

// ---------------------------------------------------------------------------
// Phase 1: one pass over W. Each block: 64 tokens x 256 vocab cols.
// bf16 hi/lo split GEMM (3 WMMAs per tile) with fp32 accumulate, then
// per-chunk online-softmax partials (row max, sum exp) + target-logit gather.
// Grid is (mtile fastest, nchunk) so co-resident blocks share W chunks in L2:
// W streams from HBM exactly once (~1.64 GB), re-reads served by L2.
// ---------------------------------------------------------------------------
__global__ __launch_bounds__(256)
void lmhead_partial_kernel(const float* __restrict__ X, const float* __restrict__ W,
                           const int* __restrict__ labels,
                           float* __restrict__ part_m, float* __restrict__ part_s,
                           float* __restrict__ tgt) {
  __shared__ unsigned short lds_w_hi[BN * KP];  // W tile transposed: [n][k]
  __shared__ unsigned short lds_w_lo[BN * KP];
  __shared__ unsigned short lds_a_hi[BM * KP];  // X tile: [m][k]
  __shared__ unsigned short lds_a_lo[BM * KP];
  __shared__ float red_m[BM][4];
  __shared__ float red_s[BM][4];

  const int t      = threadIdx.x;
  const int mtile  = blockIdx.x;   // fastest-varying: blocks sharing nchunk co-resident
  const int nchunk = blockIdx.y;
  const int n0     = nchunk * BN;
  const int m0     = mtile * BM;

  const int wave  = t >> 5;
  const int lane  = t & 31;
  const int msub0 = wave >> 2;   // strips msub0 and msub0+2
  const int nq    = wave & 3;    // 0..3 : which 64-column strip
  const int lhalf = lane >> 4;   // 0..1
  const int l16   = lane & 15;

  v8f acc00 = {}, acc01 = {}, acc02 = {}, acc03 = {};  // strip msub0
  v8f acc10 = {}, acc11 = {}, acc12 = {}, acc13 = {};  // strip msub0+2

  for (int ks = 0; ks < KSTEPS; ++ks) {
    const int k0 = ks * BK;

    // ---- stage X tile: 64x32 fp32 -> bf16 hi/lo, packed b128 stores ----
    {
      const int am = t >> 2;          // 0..63
      const int ak = (t & 3) * 8;     // 0,8,16,24
      const float4 va = *(const float4*)&X[(size_t)(m0 + am) * D_ + k0 + ak];
      const float4 vb = *(const float4*)&X[(size_t)(m0 + am) * D_ + k0 + ak + 4];
      ushort8 hi, lo;
#pragma unroll
      for (int i = 0; i < 4; ++i) {
        unsigned h, l;
        split_bf16((&va.x)[i], h, l);
        hi[i] = (unsigned short)h; lo[i] = (unsigned short)l;
      }
#pragma unroll
      for (int i = 0; i < 4; ++i) {
        unsigned h, l;
        split_bf16((&vb.x)[i], h, l);
        hi[4 + i] = (unsigned short)h; lo[4 + i] = (unsigned short)l;
      }
      *(ushort8*)&lds_a_hi[am * KP + ak] = hi;
      *(ushort8*)&lds_a_lo[am * KP + ak] = lo;
    }

    // ---- stage W tile: 32(K) x 256(N) fp32 -> transposed bf16 hi/lo [n][k] ----
    // Each thread handles K-pairs (k, k+1) for 4 consecutive n, so the two
    // bf16 halves pack into one dword -> ds_store_b32 (half the store count).
    {
      const int wn  = (t & 63) * 4;   // 0..252
      const int kb2 = (t >> 6) * 2;   // 0,2,4,6
#pragma unroll
      for (int p = 0; p < 4; ++p) {
        const int k = kb2 + p * 8;    // covers 0..30 in pairs
        const float4 v0 = *(const float4*)&W[(size_t)(k0 + k) * V_ + n0 + wn];
        const float4 v1 = *(const float4*)&W[(size_t)(k0 + k + 1) * V_ + n0 + wn];
        if (k0 + BK < D_) {  // prefetch next K tile (global_prefetch_b8, no counter)
          __builtin_prefetch(&W[(size_t)(k0 + BK + k) * V_ + n0 + wn], 0, 3);
          __builtin_prefetch(&W[(size_t)(k0 + BK + k + 1) * V_ + n0 + wn], 0, 3);
        }
#pragma unroll
        for (int i = 0; i < 4; ++i) {
          unsigned h0, l0, h1, l1;
          split_bf16((&v0.x)[i], h0, l0);
          split_bf16((&v1.x)[i], h1, l1);
          *(unsigned*)&lds_w_hi[(wn + i) * KP + k] = h0 | (h1 << 16);
          *(unsigned*)&lds_w_lo[(wn + i) * KP + k] = l0 | (l1 << 16);
        }
      }
    }
    __syncthreads();

    // ---- operands: A per strip (hi/lo), B shared across both strips ----
    // A 16x32 bf16 per-lane layout: row = l16, K = {b..b+7, b+16..b+23}, b = 8*lhalf
    BfPack a0_hi, a0_lo, a1_hi, a1_lo;
    {
      const int r0 = (msub0 * 16 + l16) * KP + lhalf * 8;
      a0_hi.h[0] = *(const ushort8*)&lds_a_hi[r0];
      a0_hi.h[1] = *(const ushort8*)&lds_a_hi[r0 + 16];
      a0_lo.h[0] = *(const ushort8*)&lds_a_lo[r0];
      a0_lo.h[1] = *(const ushort8*)&lds_a_lo[r0 + 16];
      const int r1 = ((msub0 + 2) * 16 + l16) * KP + lhalf * 8;
      a1_hi.h[0] = *(const ushort8*)&lds_a_hi[r1];
      a1_hi.h[1] = *(const ushort8*)&lds_a_hi[r1 + 16];
      a1_lo.h[0] = *(const ushort8*)&lds_a_lo[r1];
      a1_lo.h[1] = *(const ushort8*)&lds_a_lo[r1 + 16];
    }

    // B 32x16 bf16 per-lane layout: N = l16, K = {b..b+7, b+16..b+23}, b = 8*lhalf
    const int bkb = lhalf * 8;
#define TILE_J(J, ACCA, ACCB)                                                          \
    {                                                                                  \
      const int bo = (nq * 64 + (J) * 16 + l16) * KP + bkb;                            \
      BfPack b_hi, b_lo;                                                               \
      b_hi.h[0] = *(const ushort8*)&lds_w_hi[bo];                                      \
      b_hi.h[1] = *(const ushort8*)&lds_w_hi[bo + 16];                                 \
      b_lo.h[0] = *(const ushort8*)&lds_w_lo[bo];                                      \
      b_lo.h[1] = *(const ushort8*)&lds_w_lo[bo + 16];                                 \
      ACCA = __builtin_amdgcn_wmma_f32_16x16x32_bf16(false, a0_hi.bf, false, b_hi.bf, (short)0, ACCA, false, false); \
      ACCA = __builtin_amdgcn_wmma_f32_16x16x32_bf16(false, a0_hi.bf, false, b_lo.bf, (short)0, ACCA, false, false); \
      ACCA = __builtin_amdgcn_wmma_f32_16x16x32_bf16(false, a0_lo.bf, false, b_hi.bf, (short)0, ACCA, false, false); \
      ACCB = __builtin_amdgcn_wmma_f32_16x16x32_bf16(false, a1_hi.bf, false, b_hi.bf, (short)0, ACCB, false, false); \
      ACCB = __builtin_amdgcn_wmma_f32_16x16x32_bf16(false, a1_hi.bf, false, b_lo.bf, (short)0, ACCB, false, false); \
      ACCB = __builtin_amdgcn_wmma_f32_16x16x32_bf16(false, a1_lo.bf, false, b_hi.bf, (short)0, ACCB, false, false); \
    }
    TILE_J(0, acc00, acc10)
    TILE_J(1, acc01, acc11)
    TILE_J(2, acc02, acc12)
    TILE_J(3, acc03, acc13)
#undef TILE_J
    __syncthreads();
  }

  // ---- epilogue: online-softmax partials per strip, then cross-wave combine ----
  strip_epilogue(acc00, acc01, acc02, acc03, msub0,     nq, lhalf, l16, m0, n0,
                 labels, red_m, red_s, tgt);
  strip_epilogue(acc10, acc11, acc12, acc13, msub0 + 2, nq, lhalf, l16, m0, n0,
                 labels, red_m, red_s, tgt);
  __syncthreads();

  if (t < BM) {
    const float m4 = fmaxf(fmaxf(red_m[t][0], red_m[t][1]), fmaxf(red_m[t][2], red_m[t][3]));
    const float s4 = red_s[t][0] * __expf(red_m[t][0] - m4)
                   + red_s[t][1] * __expf(red_m[t][1] - m4)
                   + red_s[t][2] * __expf(red_m[t][2] - m4)
                   + red_s[t][3] * __expf(red_m[t][3] - m4);
    const int token = m0 + t;
    part_m[(size_t)token * NCHUNK + nchunk] = m4;
    part_s[(size_t)token * NCHUNK + nchunk] = s4;
  }
}

// ---------------------------------------------------------------------------
// Phase 2: per-token combine of 392 chunk partials -> lse -> ce/z per token.
// ---------------------------------------------------------------------------
__global__ __launch_bounds__(256)
void token_reduce_kernel(const float* __restrict__ part_m, const float* __restrict__ part_s,
                         const float* __restrict__ tgt, const int* __restrict__ labels,
                         float* __restrict__ tok_ce, float* __restrict__ tok_z) {
  __shared__ float sm[256];
  const int t   = threadIdx.x;
  const int tok = blockIdx.x;
  const float* pm = part_m + (size_t)tok * NCHUNK;
  const float* ps = part_s + (size_t)tok * NCHUNK;

  float mx = -INFINITY;
  for (int c = t; c < NCHUNK; c += 256) mx = fmaxf(mx, pm[c]);
  sm[t] = mx; __syncthreads();
  for (int off = 128; off > 0; off >>= 1) { if (t < off) sm[t] = fmaxf(sm[t], sm[t + off]); __syncthreads(); }
  const float M = sm[0]; __syncthreads();

  float s = 0.f;
  for (int c = t; c < NCHUNK; c += 256) s += ps[c] * __expf(pm[c] - M);
  sm[t] = s; __syncthreads();
  for (int off = 128; off > 0; off >>= 1) { if (t < off) sm[t] += sm[t + off]; __syncthreads(); }

  if (t == 0) {
    const float lse = M + __logf(sm[0]);
    const int lbl = labels[tok];
    const float valid = (lbl != IGNORE_INDEX) ? 1.f : 0.f;
    const float tv = (lbl != IGNORE_INDEX) ? tgt[tok] : 0.f;
    tok_ce[tok] = valid * (lse - tv);
    tok_z[tok]  = Z_MULT * valid * lse * lse;
  }
}

// ---------------------------------------------------------------------------
// Phase 3: deterministic fixed-order sum over tokens -> d_out[2].
// ---------------------------------------------------------------------------
__global__ __launch_bounds__(256)
void final_reduce_kernel(const float* __restrict__ tok_ce, const float* __restrict__ tok_z,
                         float* __restrict__ out) {
  __shared__ float s1[256];
  __shared__ float s2[256];
  const int t = threadIdx.x;
  float a = 0.f, b = 0.f;
  for (int i = t; i < BT_; i += 256) { a += tok_ce[i]; b += tok_z[i]; }
  s1[t] = a; s2[t] = b; __syncthreads();
  for (int off = 128; off > 0; off >>= 1) {
    if (t < off) { s1[t] += s1[t + off]; s2[t] += s2[t + off]; }
    __syncthreads();
  }
  if (t == 0) { out[0] = s1[0]; out[1] = s2[0]; }
}

extern "C" void kernel_launch(void* const* d_in, const int* in_sizes, int n_in,
                              void* d_out, int out_size, void* d_ws, size_t ws_size,
                              hipStream_t stream) {
  (void)in_sizes; (void)n_in; (void)out_size; (void)ws_size;
  const float* X      = (const float*)d_in[0];
  const float* W      = (const float*)d_in[1];
  const int*   labels = (const int*)d_in[2];
  float* out = (float*)d_out;

  float* ws     = (float*)d_ws;
  float* part_m = ws;
  float* part_s = part_m + (size_t)BT_ * NCHUNK;
  float* tgt    = part_s + (size_t)BT_ * NCHUNK;
  float* tok_ce = tgt + BT_;
  float* tok_z  = tok_ce + BT_;

  dim3 g1(MTILES, NCHUNK);  // mtile fastest => co-resident blocks share W chunks in L2
  lmhead_partial_kernel<<<g1, 256, 0, stream>>>(X, W, labels, part_m, part_s, tgt);
  token_reduce_kernel<<<BT_, 256, 0, stream>>>(part_m, part_s, tgt, labels, tok_ce, tok_z);
  final_reduce_kernel<<<1, 256, 0, stream>>>(tok_ce, tok_z, out);
}
